// MultiHeadGroupedQueryAttention_83459804496303
// MI455X (gfx1250) — compile-verified, matrix operand order checked by reading, not run on hardware
//
#include <hip/hip_runtime.h>
#include <hip/hip_bf16.h>

// ---------------------------------------------------------------------------
// Problem constants (match reference)
// ---------------------------------------------------------------------------
constexpr int MODEL_DIM = 4096;
constexpr int HEAD_DIM  = 128;
constexpr int N_HEADS   = 32;
constexpr int N_KV      = 8;
constexpr int BATCH     = 2;
constexpr int SEQ       = 2048;
constexpr int ROWS      = BATCH * SEQ;          // 4096 token rows
constexpr int KV_DIM    = HEAD_DIM * N_KV;      // 1024

typedef __attribute__((ext_vector_type(16))) __bf16 v16bf;
typedef __attribute__((ext_vector_type(8)))  float  v8f;

union Frag { v16bf v; unsigned u[8]; };

__device__ __forceinline__ unsigned short f2bf(float f) {
    unsigned u = __float_as_uint(f);
    u = (u + 0x7FFFu + ((u >> 16) & 1u)) >> 16;   // round-to-nearest-even
    return (unsigned short)u;
}
__device__ __forceinline__ float bf2f(unsigned short h) {
    return __uint_as_float(((unsigned)h) << 16);
}

// ---------------------------------------------------------------------------
// f32 -> bf16 conversion
// ---------------------------------------------------------------------------
__global__ void cvt_bf16_kernel(const float* __restrict__ in,
                                unsigned short* __restrict__ out, size_t n) {
    size_t i = (size_t)blockIdx.x * blockDim.x + threadIdx.x;
    if (i < n) out[i] = f2bf(in[i]);
}

// ---------------------------------------------------------------------------
// Tiled WMMA GEMM:  C[M,N] = A[M,K] * W[N,K]^T   (A,W bf16 raw u16)
// Block = 256 threads (8 waves). Tile 128x128, BK = 32.
// Waves arranged 4 (M) x 2 (N); each wave computes a 32x64 block
// (2 A-frags x 4 B-frags -> 8 WMMAs per K-step).
// Staging: double-buffered GLOBAL_LOAD_ASYNC_TO_LDS_B128 (ASYNCcnt),
// so the DMA of tile k+1 overlaps the WMMAs on tile k.
// ---------------------------------------------------------------------------
template<bool BF16OUT>
__global__ void __launch_bounds__(256)
gemm_xwT_kernel(const unsigned short* __restrict__ A,
                const unsigned short* __restrict__ W,
                void* __restrict__ Cout, int M, int N, int K) {
    __shared__ __align__(16) unsigned short As[2][128 * 32];
    __shared__ __align__(16) unsigned short Ws[2][128 * 32];

    const int n0 = blockIdx.x * 128;
    const int m0 = blockIdx.y * 128;
    const int t    = threadIdx.x;
    const int w    = t >> 5;
    const int ln   = t & 31;
    const int half = ln >> 4;
    const int l16  = ln & 15;
    const int wm   = w & 3;   // M sub-tile (0..3), 32 rows each
    const int wn   = w >> 2;  // N sub-tile (0..1), 64 cols each

    v8f acc[2][4];
#pragma unroll
    for (int sm = 0; sm < 2; ++sm)
#pragma unroll
        for (int nt = 0; nt < 4; ++nt) acc[sm][nt] = v8f{};

    // Issue async global->LDS copies for one 128x32 A tile + W tile.
    // Each tile = 512 uint4; 256 threads -> 2 per thread per tile.
    auto stage = [&](int bufi, int kt) {
        unsigned as_base = (unsigned)(uintptr_t)(&As[bufi][0]);
        unsigned ws_base = (unsigned)(uintptr_t)(&Ws[bufi][0]);
#pragma unroll
        for (int i = 0; i < 2; ++i) {
            int c   = t * 2 + i;
            int row = c >> 2;      // 0..127
            int q4  = c & 3;       // uint4 within the 32-element row
            unsigned long long ga =
                (unsigned long long)(uintptr_t)(A + (size_t)(m0 + row) * K + kt) +
                (unsigned long long)(q4 * 16);
            asm volatile("global_load_async_to_lds_b128 %0, %1, off"
                         :: "v"(as_base + (unsigned)(c * 16)), "v"(ga) : "memory");
            unsigned long long gw =
                (unsigned long long)(uintptr_t)(W + (size_t)(n0 + row) * K + kt) +
                (unsigned long long)(q4 * 16);
            asm volatile("global_load_async_to_lds_b128 %0, %1, off"
                         :: "v"(ws_base + (unsigned)(c * 16)), "v"(gw) : "memory");
        }
    };

    stage(0, 0);
    int buf = 0;
    for (int kt = 0; kt < K; kt += 32) {
        asm volatile("s_wait_asynccnt 0x0" ::: "memory");
        __syncthreads();
        if (kt + 32 < K) stage(buf ^ 1, kt + 32);

        const unsigned* Au = (const unsigned*)As[buf];
        const unsigned* Wu = (const unsigned*)Ws[buf];

        Frag a[2];
#pragma unroll
        for (int sm = 0; sm < 2; ++sm)
#pragma unroll
            for (int p = 0; p < 8; ++p)
                a[sm].u[p] = Au[(32 * wm + 16 * sm + l16) * 16 +
                                (p >> 2) * 8 + half * 4 + (p & 3)];

#pragma unroll
        for (int nt = 0; nt < 4; ++nt) {
            Frag b;
#pragma unroll
            for (int p = 0; p < 8; ++p)
                b.u[p] = Wu[(64 * wn + 16 * nt + l16) * 16 +
                            (p >> 2) * 8 + half * 4 + (p & 3)];
#pragma unroll
            for (int sm = 0; sm < 2; ++sm)
                acc[sm][nt] = __builtin_amdgcn_wmma_f32_16x16x32_bf16(
                    false, a[sm].v, false, b.v, (short)0, acc[sm][nt], false, false);
        }
        buf ^= 1;
    }

#pragma unroll
    for (int sm = 0; sm < 2; ++sm)
#pragma unroll
        for (int nt = 0; nt < 4; ++nt)
#pragma unroll
            for (int r = 0; r < 8; ++r) {
                int row = m0 + 32 * wm + 16 * sm + r + 8 * half;
                int col = n0 + 64 * wn + 16 * nt + l16;
                float val = acc[sm][nt][r];
                if (BF16OUT)
                    ((unsigned short*)Cout)[(size_t)row * N + col] = f2bf(val);
                else
                    ((float*)Cout)[(size_t)row * N + col] = val;
            }
}

// ---------------------------------------------------------------------------
// In-place interleaved-pair RoPE on bf16 [ROWS][nheads*HEAD_DIM]
// ---------------------------------------------------------------------------
__global__ void rope_kernel(unsigned short* __restrict__ qk,
                            const int* __restrict__ posp, int nheads) {
    const int halfd = HEAD_DIM / 2;  // 64
    size_t idx = (size_t)blockIdx.x * blockDim.x + threadIdx.x;
    size_t total = (size_t)ROWS * nheads * halfd;
    if (idx >= total) return;
    int    i   = (int)(idx % halfd);
    size_t rem = idx / halfd;
    int    h   = (int)(rem % nheads);
    size_t row = rem / nheads;                 // b*SEQ + s
    int    s   = (int)(row % SEQ);
    float pos = (float)(*posp) + (float)s;
    float inv_freq = __powf(10000.0f, -((float)(2 * i)) / (float)HEAD_DIM);
    float ang = pos * inv_freq;
    float c, sn;
    __sincosf(ang, &sn, &c);
    size_t base = row * (size_t)(nheads * HEAD_DIM) + (size_t)h * HEAD_DIM + 2 * i;
    float x1 = bf2f(qk[base]);
    float x2 = bf2f(qk[base + 1]);
    qk[base]     = f2bf(x1 * c - x2 * sn);
    qk[base + 1] = f2bf(x1 * sn + x2 * c);
}

// ---------------------------------------------------------------------------
// Flash-attention (causal, GQA). Block = 128 threads (4 waves).
// Grid (SEQ/64, N_HEADS, BATCH). Each block: 64 query rows of one head.
// ---------------------------------------------------------------------------
__global__ void __launch_bounds__(128)
attn_kernel(const unsigned short* __restrict__ Q,
            const unsigned short* __restrict__ Kb,
            const unsigned short* __restrict__ Vb,
            unsigned short* __restrict__ O) {
    __shared__ __align__(16) unsigned short Qs[64 * 128];
    __shared__ __align__(16) unsigned short Ks[64 * 128];
    __shared__ __align__(16) unsigned short Vt[128 * 64];   // transposed: [d][key]
    __shared__ __align__(16) unsigned short Ps[4 * 16 * 64];

    const int qt  = blockIdx.x;
    const int h   = blockIdx.y;
    const int b   = blockIdx.z;
    const int kvh = h >> 2;
    const int t    = threadIdx.x;
    const int w    = t >> 5;
    const int ln   = t & 31;
    const int half = ln >> 4;
    const int l16  = ln & 15;

    // ---- stage Q tile [64][128] ----
#pragma unroll
    for (int i = 0; i < 8; ++i) {
        int c = t * 8 + i;
        int row = c >> 4;
        int q4  = c & 15;
        ((uint4*)Qs)[c] =
            *((const uint4*)(Q + (size_t)(b * SEQ + qt * 64 + row) * MODEL_DIM + h * HEAD_DIM) + q4);
    }
    __syncthreads();

    unsigned* Qs_u = (unsigned*)Qs;
    unsigned* Ks_u = (unsigned*)Ks;
    unsigned* Vt_u = (unsigned*)Vt;
    unsigned* Ps_u = (unsigned*)Ps;

    // Preload Q A-fragments (4 k-steps of 32 over HEAD_DIM=128)
    Frag aq[4];
#pragma unroll
    for (int kk = 0; kk < 4; ++kk)
#pragma unroll
        for (int p = 0; p < 8; ++p)
            aq[kk].u[p] = Qs_u[(16 * w + l16) * 64 + kk * 16 + (p >> 2) * 8 + half * 4 + (p & 3)];

    v8f acco[8];
#pragma unroll
    for (int ot = 0; ot < 8; ++ot) acco[ot] = v8f{};
    float mrow[8], lrow[8];
#pragma unroll
    for (int r = 0; r < 8; ++r) { mrow[r] = -1e30f; lrow[r] = 0.0f; }
    const float scale = 0.08838834764831845f;   // 1/sqrt(128)

    for (int jt = 0; jt <= qt; ++jt) {
        __syncthreads();
        // ---- stage K tile [64][128] and V tile transposed [128][64] ----
#pragma unroll
        for (int i = 0; i < 8; ++i) {
            int c = t * 8 + i;
            int row = c >> 4;
            int q4  = c & 15;
            size_t gbase = (size_t)(b * SEQ + jt * 64 + row) * KV_DIM + kvh * HEAD_DIM;
            ((uint4*)Ks)[c] = *((const uint4*)(Kb + gbase) + q4);
            uint4 vv = *((const uint4*)(Vb + gbase) + q4);
            unsigned short* vs = (unsigned short*)&vv;
#pragma unroll
            for (int j = 0; j < 8; ++j) Vt[(q4 * 8 + j) * 64 + row] = vs[j];
        }
        __syncthreads();

        // ---- scores: S = Q * K^T, 16x64 strip per wave ----
        v8f accs[4];
#pragma unroll
        for (int nt = 0; nt < 4; ++nt) accs[nt] = v8f{};
#pragma unroll
        for (int nt = 0; nt < 4; ++nt)
#pragma unroll
            for (int kk = 0; kk < 4; ++kk) {
                Frag bk;
#pragma unroll
                for (int p = 0; p < 8; ++p)
                    bk.u[p] = Ks_u[(nt * 16 + l16) * 64 + kk * 16 + (p >> 2) * 8 + half * 4 + (p & 3)];
                accs[nt] = __builtin_amdgcn_wmma_f32_16x16x32_bf16(
                    false, aq[kk].v, false, bk.v, (short)0, accs[nt], false, false);
            }

        // ---- causal mask + online softmax ----
        const int qrow_base = qt * 64 + 16 * w + 8 * half;
#pragma unroll
        for (int r = 0; r < 8; ++r) {
            int qrow = qrow_base + r;
            float sv[4];
            float mloc = -1e30f;
#pragma unroll
            for (int nt = 0; nt < 4; ++nt) {
                int key = jt * 64 + nt * 16 + l16;
                float s = accs[nt][r] * scale;
                if (key > qrow) s = -1e30f;
                sv[nt] = s;
                mloc = fmaxf(mloc, s);
            }
            // row max across the 16-lane group (C layout: one row spans 16 lanes)
            for (int msk = 8; msk >= 1; msk >>= 1)
                mloc = fmaxf(mloc, __shfl_xor(mloc, msk, 32));
            float mnew = fmaxf(mrow[r], mloc);
            float corr = __expf(mrow[r] - mnew);
            float rs = 0.0f;
#pragma unroll
            for (int nt = 0; nt < 4; ++nt) {
                float p = __expf(sv[nt] - mnew);
                accs[nt][r] = p;
                rs += p;
            }
            for (int msk = 8; msk >= 1; msk >>= 1)
                rs += __shfl_xor(rs, msk, 32);
            lrow[r] = lrow[r] * corr + rs;
            mrow[r] = mnew;
#pragma unroll
            for (int ot = 0; ot < 8; ++ot) acco[ot][r] *= corr;
        }

        // ---- restage P (C layout -> A layout) through LDS ----
#pragma unroll
        for (int nt = 0; nt < 4; ++nt)
#pragma unroll
            for (int r = 0; r < 8; ++r)
                Ps[(w * 16 + r + 8 * half) * 64 + nt * 16 + l16] = f2bf(accs[nt][r]);
        __syncthreads();

        // ---- O += P * V ----
#pragma unroll
        for (int ot = 0; ot < 8; ++ot)
#pragma unroll
            for (int kk = 0; kk < 2; ++kk) {
                Frag ap, bv;
#pragma unroll
                for (int p = 0; p < 8; ++p) {
                    ap.u[p] = Ps_u[(w * 16 + l16) * 32 + kk * 16 + (p >> 2) * 8 + half * 4 + (p & 3)];
                    bv.u[p] = Vt_u[(ot * 16 + l16) * 32 + kk * 16 + (p >> 2) * 8 + half * 4 + (p & 3)];
                }
                acco[ot] = __builtin_amdgcn_wmma_f32_16x16x32_bf16(
                    false, ap.v, false, bv.v, (short)0, acco[ot], false, false);
            }
    }

    // ---- normalize and write attention output (bf16, [b,s,h,d] flat) ----
#pragma unroll
    for (int r = 0; r < 8; ++r) {
        float inv = 1.0f / lrow[r];
        int srow = qt * 64 + 16 * w + r + 8 * half;
        size_t rowg = (size_t)(b * SEQ + srow) * MODEL_DIM + (size_t)h * HEAD_DIM;
#pragma unroll
        for (int ot = 0; ot < 8; ++ot)
            O[rowg + ot * 16 + l16] = f2bf(acco[ot][r] * inv);
    }
}

// ---------------------------------------------------------------------------
// Host-side orchestration
// ---------------------------------------------------------------------------
extern "C" void kernel_launch(void* const* d_in, const int* in_sizes, int n_in,
                              void* d_out, int out_size, void* d_ws, size_t ws_size,
                              hipStream_t stream) {
    const float* x    = (const float*)d_in[0];
    const float* wq   = (const float*)d_in[1];
    const float* wk   = (const float*)d_in[2];
    const float* wv   = (const float*)d_in[3];
    const float* wo   = (const float*)d_in[4];
    const int*   posp = (const int*)d_in[6];
    float* out = (float*)d_out;

    // Workspace carve-out (bf16 buffers), 256B aligned bumps.
    char* base = (char*)d_ws;
    size_t off = 0;
    auto alloc = [&](size_t bytes) -> unsigned short* {
        unsigned short* p = (unsigned short*)(base + off);
        off += (bytes + 255) & ~(size_t)255;
        return p;
    };
    unsigned short* xb  = alloc((size_t)ROWS * MODEL_DIM * 2);
    unsigned short* wqb = alloc((size_t)MODEL_DIM * MODEL_DIM * 2);
    unsigned short* wkb = alloc((size_t)KV_DIM * MODEL_DIM * 2);
    unsigned short* wvb = alloc((size_t)KV_DIM * MODEL_DIM * 2);
    unsigned short* wob = alloc((size_t)MODEL_DIM * MODEL_DIM * 2);
    unsigned short* qb  = alloc((size_t)ROWS * MODEL_DIM * 2);
    unsigned short* kb  = alloc((size_t)ROWS * KV_DIM * 2);
    unsigned short* vb  = alloc((size_t)ROWS * KV_DIM * 2);
    unsigned short* aob = alloc((size_t)ROWS * MODEL_DIM * 2);
    (void)ws_size; (void)in_sizes; (void)n_in; (void)out_size;

    auto cvt = [&](const float* src, unsigned short* dst, size_t n) {
        int blocks = (int)((n + 255) / 256);
        cvt_bf16_kernel<<<blocks, 256, 0, stream>>>(src, dst, n);
    };
    cvt(x,  xb,  (size_t)ROWS * MODEL_DIM);
    cvt(wq, wqb, (size_t)MODEL_DIM * MODEL_DIM);
    cvt(wk, wkb, (size_t)KV_DIM * MODEL_DIM);
    cvt(wv, wvb, (size_t)KV_DIM * MODEL_DIM);
    cvt(wo, wob, (size_t)MODEL_DIM * MODEL_DIM);

    // Projections (bf16 output)
    gemm_xwT_kernel<true><<<dim3(MODEL_DIM / 128, ROWS / 128), 256, 0, stream>>>(
        xb, wqb, qb, ROWS, MODEL_DIM, MODEL_DIM);
    gemm_xwT_kernel<true><<<dim3(KV_DIM / 128, ROWS / 128), 256, 0, stream>>>(
        xb, wkb, kb, ROWS, KV_DIM, MODEL_DIM);
    gemm_xwT_kernel<true><<<dim3(KV_DIM / 128, ROWS / 128), 256, 0, stream>>>(
        xb, wvb, vb, ROWS, KV_DIM, MODEL_DIM);

    // RoPE (in place on bf16 Q and K)
    {
        size_t nq = (size_t)ROWS * N_HEADS * (HEAD_DIM / 2);
        rope_kernel<<<(int)((nq + 255) / 256), 256, 0, stream>>>(qb, posp, N_HEADS);
        size_t nk = (size_t)ROWS * N_KV * (HEAD_DIM / 2);
        rope_kernel<<<(int)((nk + 255) / 256), 256, 0, stream>>>(kb, posp, N_KV);
    }

    // Flash attention
    attn_kernel<<<dim3(SEQ / 64, N_HEADS, BATCH), 128, 0, stream>>>(qb, kb, vb, aob);

    // Output projection (f32 output into d_out)
    gemm_xwT_kernel<false><<<dim3(MODEL_DIM / 128, ROWS / 128), 256, 0, stream>>>(
        aob, wob, out, ROWS, MODEL_DIM, MODEL_DIM);
}
